// ConvDY2d_27281632264774
// MI455X (gfx1250) — compile-verified, matrix operand order checked by reading, not run on hardware
//
#include <hip/hip_runtime.h>
#include <hip/hip_bf16.h>

// ---------------------------------------------------------------------------
// Dynamic conv (CondConv) for MI455X / gfx1250, wave32 + WMMA bf16 + TDM.
// B=16, C=256, H=W=64, O=256, K=4, KS=3, MID=64
// Conv as implicit GEMM per sample: M=O=256, N=HW=4096, K=C*9=2304 (bf16 in,
// fp32 acc). Reduction ordered (kh, c-chunk, kw): one 66x32 halo'd row tile
// feeds 3 kw positions. Weights DMA'd by the Tensor Data Mover; both LDS
// tiles double-buffered -> ONE barrier per reduction chunk.
// ---------------------------------------------------------------------------

typedef __attribute__((ext_vector_type(16))) __bf16 v16bf;
typedef __attribute__((ext_vector_type(8)))  float  v8f;

#define BB   16
#define CC   256
#define HH   64
#define WW   64
#define OO   256
#define MID  64
#define RED  (CC * 9)          // 2304
#define HWSZ (HH * WW)         // 4096
#define XSTR 40                // Xs row stride in bf16 (80 B, 16B aligned)
#define WSTR 104               // Ws row stride in bf16 (208 B, 16B aligned)
#define NCHUNK 24              // 3 kh * 8 c-chunks of 32

#if __has_builtin(__builtin_amdgcn_tensor_load_to_lds) && \
    __has_builtin(__builtin_amdgcn_s_wait_tensorcnt)
#define USE_TDM 1
#else
#define USE_TDM 0
#endif

// ---------------------------------------------------------------------------
// K0: global average pool + fp32 -> bf16 conversion of x (single pass)
// ---------------------------------------------------------------------------
__global__ __launch_bounds__(256) void k_pool_cvt(const float* __restrict__ x,
                                                  float* __restrict__ pooled,
                                                  __bf16* __restrict__ xbf) {
    int bc = blockIdx.x;                       // b*C + c
    const float* p = x + (size_t)bc * HWSZ;
    __bf16* q = xbf + (size_t)bc * HWSZ;
    float s = 0.f;
    for (int i = threadIdx.x; i < HWSZ; i += 256) {
        float v = p[i];
        q[i] = (__bf16)v;
        s += v;
    }
    __shared__ float red[256];
    red[threadIdx.x] = s;
    __syncthreads();
    for (int off = 128; off > 0; off >>= 1) {
        if (threadIdx.x < off) red[threadIdx.x] += red[threadIdx.x + off];
        __syncthreads();
    }
    if (threadIdx.x == 0) pooled[bc] = red[0] * (1.0f / (float)HWSZ);
}

// ---------------------------------------------------------------------------
// K1: MLP + softmax attention (tiny). One block of 256 threads.
// ---------------------------------------------------------------------------
__global__ __launch_bounds__(256) void k_attn(const float* __restrict__ pooled,
                                              const float* __restrict__ fc1w,
                                              const float* __restrict__ fc1b,
                                              const float* __restrict__ fc2w,
                                              const float* __restrict__ fc2b,
                                              float* __restrict__ attn) {
    __shared__ float P[BB * CC];
    __shared__ float Hd[BB * MID];
    __shared__ float Lg[BB * 4];
    for (int i = threadIdx.x; i < BB * CC; i += 256) P[i] = pooled[i];
    __syncthreads();
    for (int e = threadIdx.x; e < BB * MID; e += 256) {
        int b = e >> 6, m = e & 63;
        float s = fc1b[m];
        for (int j = 0; j < CC; ++j) s += P[b * CC + j] * fc1w[m * CC + j];
        Hd[e] = s > 0.f ? s : 0.f;
    }
    __syncthreads();
    if (threadIdx.x < BB * 4) {
        int b = threadIdx.x >> 2, k = threadIdx.x & 3;
        float s = fc2b[k];
        for (int j = 0; j < MID; ++j) s += Hd[b * MID + j] * fc2w[k * MID + j];
        Lg[threadIdx.x] = s * (1.0f / 30.0f);
    }
    __syncthreads();
    if (threadIdx.x < BB) {
        int b = threadIdx.x;
        float mx = Lg[b * 4];
        for (int k = 1; k < 4; ++k) mx = fmaxf(mx, Lg[b * 4 + k]);
        float e0 = __expf(Lg[b * 4 + 0] - mx);
        float e1 = __expf(Lg[b * 4 + 1] - mx);
        float e2 = __expf(Lg[b * 4 + 2] - mx);
        float e3 = __expf(Lg[b * 4 + 3] - mx);
        float inv = 1.0f / (e0 + e1 + e2 + e3);
        attn[b * 4 + 0] = e0 * inv;
        attn[b * 4 + 1] = e1 * inv;
        attn[b * 4 + 2] = e2 * inv;
        attn[b * 4 + 3] = e3 * inv;
    }
}

// ---------------------------------------------------------------------------
// K2: mix weight banks, write bf16 weights TRANSPOSED to [b][o][khw][c]
// so each (kh,kw,c-chunk) tile is contiguous per output channel.
// ---------------------------------------------------------------------------
__global__ __launch_bounds__(256) void k_mix(const float* __restrict__ wdyn,
                                             const float* __restrict__ attn,
                                             __bf16* __restrict__ wmixT) {
    __shared__ float buf[RED];              // 9 KB transpose staging
    int bo = blockIdx.x;                    // b*O + o
    int b = bo >> 8, o = bo & 255;
    float a0 = attn[b * 4 + 0];
    float a1 = attn[b * 4 + 1];
    float a2 = attn[b * 4 + 2];
    float a3 = attn[b * 4 + 3];
    const size_t KSTR = (size_t)OO * RED;
    const float* w0 = wdyn + (size_t)o * RED;
    for (int r = threadIdx.x; r < RED; r += 256) {
        float v = a0 * w0[r] + a1 * w0[KSTR + r] + a2 * w0[2 * KSTR + r] +
                  a3 * w0[3 * KSTR + r];
        int c = r / 9;
        int khw = r - c * 9;
        buf[khw * CC + c] = v;              // transpose in LDS
    }
    __syncthreads();
    __bf16* dst = wmixT + (size_t)bo * RED;
    for (int r = threadIdx.x; r < RED; r += 256) dst[r] = (__bf16)buf[r];
}

// ---------------------------------------------------------------------------
// WMMA 16-bit operand fragment loader (ISA 7.12.2 layout), 2x ds_load_b128.
// ---------------------------------------------------------------------------
__device__ inline v16bf ldfrag(const __bf16* base, int lane) {
    const __bf16* p = base + (lane & 15) * XSTR + ((lane >> 4) << 3);
    union { uint4 q[2]; v16bf v; } u;
    u.q[0] = *(const uint4*)(p);
    u.q[1] = *(const uint4*)(p + 16);
    return u.v;
}
__device__ inline v16bf ldfragW(const __bf16* base, int lane) {
    const __bf16* p = base + (lane & 15) * WSTR + ((lane >> 4) << 3);
    union { uint4 q[2]; v16bf v; } u;
    u.q[0] = *(const uint4*)(p);
    u.q[1] = *(const uint4*)(p + 16);
    return u.v;
}

#if USE_TDM
typedef unsigned int u32x4 __attribute__((ext_vector_type(4)));
typedef int          i32x4 __attribute__((ext_vector_type(4)));
typedef int          i32x8 __attribute__((ext_vector_type(8)));

// Tensor DMA: 64 rows x 32 bf16 tile, global row stride 2304 elems,
// LDS rows packed 64B + 144B pad => 208 B = WSTR*2 padded stride.
// D# per ISA ch.8: group0 {count=1, lds_addr, global_addr, type=2},
// group1 {data_size=1(2B), pad_en, pad_interval=3 (16 DW), pad_amount=35
// (36 DW), tensor_dim0=32, tensor_dim1=64, tile_dim0=32, tile_dim1=64,
// tensor_dim0_stride=2304}.
__device__ inline void tdm_load_w_tile(unsigned lds_byte, const __bf16* g) {
    unsigned long long ga = (unsigned long long)g;
    u32x4 g0;
    g0[0] = 1u;                                            // count=1
    g0[1] = lds_byte;                                      // lds_addr
    g0[2] = (unsigned)(ga & 0xFFFFFFFFu);                  // global_addr lo
    g0[3] = (unsigned)((ga >> 32) & 0x01FFFFFFu) | (2u << 30); // hi | type=2
    i32x8 g1;
    g1[0] = (1 << 16) | (1 << 20) | (3 << 22) | (35 << 25);
    g1[1] = (32 << 16);        // tensor_dim0[15:0]=32
    g1[2] = (64 << 16);        // tensor_dim0 hi=0 | tensor_dim1[15:0]=64
    g1[3] = (32 << 16);        // tensor_dim1 hi=0 | tile_dim0=32
    g1[4] = 64;                // tile_dim1=64 | tile_dim2=0
    g1[5] = 2304;              // tensor_dim0_stride lo
    g1[6] = 0;                 // stride hi | tensor_dim1_stride lo
    g1[7] = 0;
    i32x4 z4 = {0, 0, 0, 0};
#if __clang_major__ >= 23
    i32x8 z8 = {0, 0, 0, 0, 0, 0, 0, 0};
    __builtin_amdgcn_tensor_load_to_lds(g0, g1, z4, z4, z8, 0);
#else
    __builtin_amdgcn_tensor_load_to_lds(g0, g1, z4, z4, 0);
#endif
}
#endif

// ---------------------------------------------------------------------------
// K3: implicit-GEMM conv. 256 threads = 8 waves. Block tile: sample b,
// 64 output channels, one spatial row h. Wave (wm,wn): 16 ch x 32 px
// (two 16x16 fp32 accumulators). 24 chunks of K=32 channels x 3 kw reuse.
// Both LDS tiles double-buffered: ONE barrier per chunk. Passing barrier(ci)
// proves all waves finished compute(ci-1), so writing buffer !buf (x stores,
// weight DMA) needs no trailing barrier.
// ---------------------------------------------------------------------------
__global__ __launch_bounds__(256) void k_conv(const __bf16* __restrict__ xbf,
                                              const __bf16* __restrict__ wmixT,
                                              float* __restrict__ out) {
    __shared__ __align__(16) __bf16 Ws[2][64 * WSTR];  // 2 x 13312 B
    __shared__ __align__(16) __bf16 Xs[2][66 * XSTR];  // 2 x  5280 B

    int bid = blockIdx.x;
    int b  = bid >> 8;
    int og = (bid >> 6) & 3;
    int h  = bid & 63;
    int o0 = og << 6;

    int tid  = threadIdx.x;
    int lane = tid & 31;
    int wave = tid >> 5;
    int wm = wave & 3, wn = wave >> 2;

    // x staging role: 8 threads per channel, 8 contiguous pixels each
    int cl  = tid >> 3;          // channel-in-chunk 0..31
    int w0p = (tid & 7) << 3;    // pixel base 0..56

#if !USE_TDM
    int ow = tid >> 2;           // weight row o_local 0..63
    int j0 = (tid & 3) * 3;      // three 8-elem groups of the 96-wide row
    size_t wrowbase = (size_t)(b * OO + o0 + ow) * RED;
#endif
    size_t wtile0 = (size_t)(b * OO + o0) * RED;       // uniform per block

    union XV { uint4 q; __bf16 hx[8]; };

    // halo rows p=0 and p=65 are never rewritten: zero once in both buffers
    if (tid < 128) {
        int c = tid & 31;
        int p = (tid & 32) ? 65 : 0;
        Xs[tid >> 6][p * XSTR + c] = (__bf16)0.0f;
    }

    auto loadx = [&](int ci) -> XV {
        int kh = ci >> 3;
        int c0 = (ci & 7) << 5;
        int hh = h + kh - 1;
        XV r;
        if ((unsigned)hh < (unsigned)HH)
            r.q = *(const uint4*)
                &xbf[(((size_t)(b * CC + c0 + cl)) * HH + hh) * WW + w0p];
        else
            r.q = make_uint4(0u, 0u, 0u, 0u);
        return r;
    };

#if USE_TDM
    auto tdm_issue = [&](int ci, int buf) {
        int kh = ci >> 3;
        int c0 = (ci & 7) << 5;
        unsigned lbase = (unsigned)(unsigned long long)&Ws[buf][0];
#pragma unroll
        for (int kw = 0; kw < 3; ++kw)
            tdm_load_w_tile(lbase + kw * 64,
                            wmixT + wtile0 + (size_t)(kh * 3 + kw) * CC + c0);
    };
#else
    uint4 wq0, wq1, wq2;
    auto loadw = [&](int ci) {
        int kh = ci >> 3;
        int c0 = (ci & 7) << 5;
        const __bf16* p = &wmixT[wrowbase + (size_t)(kh * 3) * CC + c0];
        wq0 = *(const uint4*)(p + ((j0 + 0) >> 2) * CC + (((j0 + 0) & 3) << 3));
        wq1 = *(const uint4*)(p + ((j0 + 1) >> 2) * CC + (((j0 + 1) & 3) << 3));
        wq2 = *(const uint4*)(p + ((j0 + 2) >> 2) * CC + (((j0 + 2) & 3) << 3));
    };
#endif

    XV xv = loadx(0);
#if USE_TDM
    if (wave == 0) tdm_issue(0, 0);
#else
    loadw(0);
#endif

    v8f acc0 = {};
    v8f acc1 = {};

    for (int ci = 0; ci < NCHUNK; ++ci) {
        int buf = ci & 1;

        // ---- stage x tile for THIS chunk (transposed: row = padded pixel)
#pragma unroll
        for (int i = 0; i < 8; ++i)
            Xs[buf][(w0p + 1 + i) * XSTR + cl] = xv.hx[i];

#if !USE_TDM
        // ---- stage weight tile from regs (16B aligned ds_store_b128)
        *(uint4*)&Ws[buf][ow * WSTR + (j0 + 0) * 8] = wq0;
        *(uint4*)&Ws[buf][ow * WSTR + (j0 + 1) * 8] = wq1;
        *(uint4*)&Ws[buf][ow * WSTR + (j0 + 2) * 8] = wq2;
#endif

#if USE_TDM
        if (wave == 0) __builtin_amdgcn_s_wait_tensorcnt(0);  // Ws[buf] ready
#endif
        __syncthreads();   // the ONLY barrier per chunk

        // ---- pipeline: next chunk into the opposite buffers / registers
        if (ci + 1 < NCHUNK) {
            xv = loadx(ci + 1);
#if USE_TDM
            if (wave == 0) tdm_issue(ci + 1, buf ^ 1);  // overlaps WMMA below
#else
            loadw(ci + 1);
#endif
        }
        if (ci + 2 < NCHUNK) {
            int kh2 = (ci + 2) >> 3, c02 = ((ci + 2) & 7) << 5;
            int hh2 = h + kh2 - 1;
            if ((unsigned)hh2 < (unsigned)HH)
                __builtin_prefetch(
                    &xbf[(((size_t)(b * CC + c02 + cl)) * HH + hh2) * WW + w0p],
                    0, 1);
        }

        // ---- 3 kw positions reuse the staged x row tile (kw = LDS row shift)
#pragma unroll
        for (int kw = 0; kw < 3; ++kw) {
            v16bf af  = ldfragW(&Ws[buf][(wm * 16) * WSTR + kw * 32], lane);
            v16bf bf0 = ldfrag(&Xs[buf][(wn * 32 + kw) * XSTR], lane);
            v16bf bf1 = ldfrag(&Xs[buf][(wn * 32 + 16 + kw) * XSTR], lane);
            acc0 = __builtin_amdgcn_wmma_f32_16x16x32_bf16(
                false, af, false, bf0, (short)0, acc0, false, false);
            acc1 = __builtin_amdgcn_wmma_f32_16x16x32_bf16(
                false, af, false, bf1, (short)0, acc1, false, false);
        }
        // no trailing barrier: next iteration writes the opposite buffers
    }

    // ---- store (C/D layout: lanes 0-15 M=j, lanes 16-31 M=j+8, N=lane&15)
    int nl = lane & 15;
    int mo = (lane >> 4) << 3;
    size_t obase = (((size_t)b * OO + o0 + wm * 16 + mo) * HH + h) * WW;
#pragma unroll
    for (int j = 0; j < 8; ++j) {
        out[obase + (size_t)j * HWSZ + wn * 32 + nl]      = acc0[j];
        out[obase + (size_t)j * HWSZ + wn * 32 + 16 + nl] = acc1[j];
    }
}

// ---------------------------------------------------------------------------
// launch
// ---------------------------------------------------------------------------
extern "C" void kernel_launch(void* const* d_in, const int* in_sizes, int n_in,
                              void* d_out, int out_size, void* d_ws, size_t ws_size,
                              hipStream_t stream) {
    const float* x    = (const float*)d_in[0];
    const float* wdyn = (const float*)d_in[1];
    const float* fc1w = (const float*)d_in[2];
    const float* fc1b = (const float*)d_in[3];
    const float* fc2w = (const float*)d_in[4];
    const float* fc2b = (const float*)d_in[5];
    float* out = (float*)d_out;

    char* ws = (char*)d_ws;
    float*  attn   = (float*)(ws);                        //    256 B
    float*  pooled = (float*)(ws + 256);                  // 16 384 B
    __bf16* xbf    = (__bf16*)(ws + 16640);               // 33 554 432 B
    __bf16* wmixT  = (__bf16*)(ws + 16640 + 33554432ull); // 18 874 368 B

    k_pool_cvt<<<BB * CC, 256, 0, stream>>>(x, pooled, xbf);
    k_attn<<<1, 256, 0, stream>>>(pooled, fc1w, fc1b, fc2w, fc2b, attn);
    k_mix<<<BB * OO, 256, 0, stream>>>(wdyn, attn, wmixT);
    k_conv<<<BB * 4 * HH, 256, 0, stream>>>(xbf, wmixT, out);
}
